// AttentionModel_without_Residual_8778913153065
// MI455X (gfx1250) — matrix operand order repaired and checked by reading, not compiled, not run on hardware
//
#include <hip/hip_runtime.h>
#include <hip/hip_bf16.h>
#include <stdint.h>

typedef __bf16 bf16_t;
typedef __attribute__((ext_vector_type(16))) __bf16 bf16x16;
typedef __attribute__((ext_vector_type(8)))  __bf16 bf16x8;
typedef __attribute__((ext_vector_type(8)))  float  f32x8;

#define B_DIM   1024
#define H_DIM   1024
#define D_DIM   512
#define P_DIM   66
#define KP_DIM  96      // P padded to multiple of 32
#define T_IN    10
#define T_OUT   25
#define FOURH   4096
#define POSE_NPAD 128   // 66 padded up to multiple of 128 (block N-tile)

// ---------------------------------------------------------------------------
// Async copy of one 128x32 bf16 tile (8KB) from global to LDS.
// 256 threads x 2 chunks x 16B = 8KB. Row-major LDS layout [128][32] halves.
// Uses CDNA5 GLOBAL_LOAD_ASYNC_TO_LDS_B128 (ASYNCcnt-tracked).
// ---------------------------------------------------------------------------
__device__ __forceinline__ void async_copy_tile(const bf16_t* gbase, int ldK,
                                                int rowBlock, int kb,
                                                const bf16_t* lds_tile, int tid)
{
  const uint32_t lbase = (uint32_t)(uintptr_t)lds_tile;
#pragma unroll
  for (int j = 0; j < 2; ++j) {
    const int c   = tid + j * 256;       // chunk 0..511
    const int row = c >> 2;              // 0..127
    const int c16 = c & 3;               // which 16B of the 64B row
    const bf16_t* g = gbase + (size_t)(rowBlock + row) * ldK + kb + c16 * 8;
    const uint32_t l = lbase + (uint32_t)c * 16u;
    asm volatile("global_load_async_to_lds_b128 %0, %1, off"
                 :: "v"(l), "v"(g) : "memory");
  }
}

__device__ __forceinline__ void wait_async_and_barrier() {
  asm volatile("s_wait_asynccnt 0" ::: "memory");
  __syncthreads();
}

// ---------------------------------------------------------------------------
// WMMA fragment loads from an LDS tile [128][32] bf16, contiguous k order, same for A and B.
// A (16x32): lane l -> row l&15; K segment [16*hi,+16).
// B (32x16): lane l -> col l&15; K segment [16*hi,+16).
// ---------------------------------------------------------------------------
__device__ __forceinline__ bf16x16 lds_frag_a(const bf16_t* t, int rowBase, int lane) {
  const int m  = rowBase + (lane & 15);
  const int hi = lane >> 4;
  const bf16_t* p = t + m * 32 + hi * 16;
  bf16x8 lo = *reinterpret_cast<const bf16x8*>(p);
  bf16x8 hh = *reinterpret_cast<const bf16x8*>(p + 8);
  return __builtin_shufflevector(lo, hh, 0,1,2,3,4,5,6,7,8,9,10,11,12,13,14,15);
}

__device__ __forceinline__ bf16x16 lds_frag_b(const bf16_t* t, int colBase, int lane) {
  const int n  = colBase + (lane & 15);
  const int hi = lane >> 4;
  const bf16_t* p = t + n * 32 + hi * 16;
  bf16x8 lo = *reinterpret_cast<const bf16x8*>(p);
  bf16x8 hh = *reinterpret_cast<const bf16x8*>(p + 8);
  return __builtin_shufflevector(lo, hh, 0,1,2,3,4,5,6,7,8,9,10,11,12,13,14,15);
}

// ---------------------------------------------------------------------------
// C[M,N] = A[M,K] * W[N,K]^T (+bias[n]) (+addend[m,n]); optional f32 / bf16
// output, optional row-permute: m = o*permInner + i  ->  r = i*permOuter + o.
// Block tile 128x128, 8 waves as 2(M) x 4(N), wave tile 64x32 (8 f32x8 acc).
// A/B tiles double-buffered in LDS via async global->LDS copies.
// Requires M%128==0, N%128==0, K%32==0 (buffers pre-padded accordingly).
// ---------------------------------------------------------------------------
__global__ __launch_bounds__(256)
void gemm_bf16_kernel(const bf16_t* __restrict__ A, const bf16_t* __restrict__ W,
                      const float* __restrict__ bias, const float* addend,
                      float* outF, bf16_t* outB,
                      int M, int N, int K, int Nstore, int ldC,
                      int permInner, int permOuter)
{
  __shared__ __align__(16) bf16_t Abuf[2][128 * 32];
  __shared__ __align__(16) bf16_t Bbuf[2][128 * 32];

  const int tid   = threadIdx.x;
  const int lane  = tid & 31;
  const int wave  = tid >> 5;
  const int waveM = wave >> 2;                 // 0..1 -> 64 rows each
  const int waveN = wave & 3;                  // 0..3 -> 32 cols each
  const int mBlk  = blockIdx.x * 128;
  const int nBlk  = blockIdx.y * 128;

  const f32x8 zero = {0.f,0.f,0.f,0.f,0.f,0.f,0.f,0.f};
  f32x8 acc[4][2];
#pragma unroll
  for (int mt = 0; mt < 4; ++mt) { acc[mt][0] = zero; acc[mt][1] = zero; }

  // prologue: stage tile 0
  async_copy_tile(A, K, mBlk, 0, &Abuf[0][0], tid);
  async_copy_tile(W, K, nBlk, 0, &Bbuf[0][0], tid);
  wait_async_and_barrier();

  const int nsteps = K >> 5;
  for (int s = 0; s < nsteps; ++s) {
    const int p = s & 1;
    if (s + 1 < nsteps) {
      async_copy_tile(A, K, mBlk, (s + 1) * 32, &Abuf[1 - p][0], tid);
      async_copy_tile(W, K, nBlk, (s + 1) * 32, &Bbuf[1 - p][0], tid);
    }
    const bf16_t* As = &Abuf[p][0];
    const bf16_t* Bs = &Bbuf[p][0];
    bf16x16 bf0 = lds_frag_b(Bs, waveN * 32,      lane);
    bf16x16 bf1 = lds_frag_b(Bs, waveN * 32 + 16, lane);
#pragma unroll
    for (int mt = 0; mt < 4; ++mt) {
      bf16x16 af = lds_frag_a(As, waveM * 64 + mt * 16, lane);
      acc[mt][0] = __builtin_amdgcn_wmma_f32_16x16x32_bf16(false, af, false, bf0,
                                                           (short)0, acc[mt][0], false, false);
      acc[mt][1] = __builtin_amdgcn_wmma_f32_16x16x32_bf16(false, af, false, bf1,
                                                           (short)0, acc[mt][1], false, false);
    }
    if (s + 1 < nsteps) wait_async_and_barrier();
  }

  // epilogue: C/D layout -> (M = r + 8*(lane>>4), N = lane&15) per VGPR r
  const int hi = lane >> 4;
  const int lo = lane & 15;
#pragma unroll
  for (int mt = 0; mt < 4; ++mt) {
#pragma unroll
    for (int nt = 0; nt < 2; ++nt) {
      const int n = nBlk + waveN * 32 + nt * 16 + lo;
      if (n >= Nstore) continue;
      const float bn = bias ? bias[n] : 0.0f;
#pragma unroll
      for (int r = 0; r < 8; ++r) {
        const int m = mBlk + waveM * 64 + mt * 16 + r + hi * 8;
        long rout = (permInner > 0)
                        ? (long)(m % permInner) * permOuter + (m / permInner)
                        : (long)m;
        float v = acc[mt][nt][r] + bn;
        const long ci = rout * (long)ldC + n;
        if (addend) v += addend[ci];
        if (outF)   outF[ci] = v;
        if (outB)   outB[ci] = (bf16_t)v;
      }
    }
  }
}

// ---------------------------------------------------------------------------
// LSTM pointwise: gates G[B,4H] (torch order i,f,g,o), c fp32 in-place,
// h written as bf16 for the next recurrent GEMM / attention.
// ---------------------------------------------------------------------------
__global__ __launch_bounds__(256)
void lstm_cell_kernel(const float* __restrict__ G, float* __restrict__ c,
                      bf16_t* __restrict__ h_out)
{
  const int idx = blockIdx.x * blockDim.x + threadIdx.x;   // B*H threads
  const int b = idx >> 10;
  const int n = idx & 1023;
  const float* g = G + (size_t)b * FOURH;
  const float gi = g[n], gf = g[H_DIM + n], gg = g[2 * H_DIM + n], go = g[3 * H_DIM + n];
  const float si = 1.0f / (1.0f + __expf(-gi));
  const float sf = 1.0f / (1.0f + __expf(-gf));
  const float so = 1.0f / (1.0f + __expf(-go));
  const float tg = tanhf(gg);
  const float cn = sf * c[idx] + si * tg;
  c[idx]     = cn;
  h_out[idx] = (bf16_t)(so * tanhf(cn));
}

// ---------------------------------------------------------------------------
// Attention: one block per batch row b. cat_h = [all_h(10), h_dec, all_hz(10)],
// scores[s] = <cat_h[s,b,:], c[b,:]>, softmax over 21, atth = sum attn*cat_h.
// ---------------------------------------------------------------------------
__device__ __forceinline__ const bf16_t* cat_slot(const bf16_t* all_h, const bf16_t* hdec,
                                                  const bf16_t* all_hz, int s, int b)
{
  if (s < T_IN)  return all_h  + ((size_t)s * B_DIM + b) * H_DIM;
  if (s == T_IN) return hdec   + (size_t)b * H_DIM;
  return all_hz + ((size_t)(s - T_IN - 1) * B_DIM + b) * H_DIM;
}

__global__ __launch_bounds__(256)
void attention_kernel(const bf16_t* __restrict__ all_h, const bf16_t* __restrict__ hdec,
                      const bf16_t* __restrict__ all_hz, const float* __restrict__ c,
                      bf16_t* __restrict__ atth)
{
  const int b = blockIdx.x;
  const int tid = threadIdx.x;
  const int h0 = tid * 4;                 // 256 threads x 4 = 1024 = H
  __shared__ float red[256];
  __shared__ float sc[T_IN * 2 + 1];

  float cv[4];
#pragma unroll
  for (int j = 0; j < 4; ++j) cv[j] = c[(size_t)b * H_DIM + h0 + j];

  for (int s = 0; s < 2 * T_IN + 1; ++s) {
    const bf16_t* src = cat_slot(all_h, hdec, all_hz, s, b);
    float p = 0.f;
#pragma unroll
    for (int j = 0; j < 4; ++j) p += (float)src[h0 + j] * cv[j];
    red[tid] = p;
    __syncthreads();
    for (int off = 128; off > 0; off >>= 1) {
      if (tid < off) red[tid] += red[tid + off];
      __syncthreads();
    }
    if (tid == 0) sc[s] = red[0];
    __syncthreads();
  }

  float mx = sc[0];
  for (int s = 1; s < 2 * T_IN + 1; ++s) mx = fmaxf(mx, sc[s]);
  float w[2 * T_IN + 1];
  float denom = 0.f;
  for (int s = 0; s < 2 * T_IN + 1; ++s) { w[s] = __expf(sc[s] - mx); denom += w[s]; }
  const float inv = 1.0f / denom;

  float outv[4] = {0.f, 0.f, 0.f, 0.f};
  for (int s = 0; s < 2 * T_IN + 1; ++s) {
    const bf16_t* src = cat_slot(all_h, hdec, all_hz, s, b);
    const float ws = w[s] * inv;
#pragma unroll
    for (int j = 0; j < 4; ++j) outv[j] += ws * (float)src[h0 + j];
  }
#pragma unroll
  for (int j = 0; j < 4; ++j) atth[(size_t)b * H_DIM + h0 + j] = (bf16_t)outv[j];
}

// ---------------------------------------------------------------------------
// Small conversion / init kernels
// ---------------------------------------------------------------------------
__global__ void k_f2bf(const float* __restrict__ s, bf16_t* __restrict__ d, long n) {
  long i = (long)blockIdx.x * blockDim.x + threadIdx.x;
  if (i < n) d[i] = (bf16_t)s[i];
}
__global__ void k_pad_cols(const float* __restrict__ s, bf16_t* __restrict__ d,
                           int R, int Kin, int Kpad) {
  long i = (long)blockIdx.x * blockDim.x + threadIdx.x;
  if (i >= (long)R * Kpad) return;
  const int r = (int)(i / Kpad), k = (int)(i % Kpad);
  d[i] = (k < Kin) ? (bf16_t)s[(long)r * Kin + k] : (bf16_t)0.0f;
}
__global__ void k_pad_rows(const float* __restrict__ s, bf16_t* __restrict__ d,
                           int Rin, int Rpad, int K) {
  long i = (long)blockIdx.x * blockDim.x + threadIdx.x;
  if (i >= (long)Rpad * K) return;
  const int r = (int)(i / K);
  d[i] = (r < Rin) ? (bf16_t)s[i] : (bf16_t)0.0f;
}
__global__ void k_add(const float* __restrict__ a, const float* __restrict__ b,
                      float* __restrict__ d, long n) {
  long i = (long)blockIdx.x * blockDim.x + threadIdx.x;
  if (i < n) d[i] = a[i] + b[i];
}
__global__ void k_zero(float* __restrict__ d, long n) {
  long i = (long)blockIdx.x * blockDim.x + threadIdx.x;
  if (i < n) d[i] = 0.0f;
}

// ---------------------------------------------------------------------------
extern "C" void kernel_launch(void* const* d_in, const int* in_sizes, int n_in,
                              void* d_out, int out_size, void* d_ws, size_t ws_size,
                              hipStream_t stream)
{
  (void)in_sizes; (void)n_in; (void)out_size; (void)ws_size;

  const float* x        = (const float*)d_in[0];
  const float* z        = (const float*)d_in[1];
  const float* wf       = (const float*)d_in[2];
  const float* bf_      = (const float*)d_in[3];
  const float* enc_wih  = (const float*)d_in[4];
  const float* enc_whh  = (const float*)d_in[5];
  const float* enc_bih  = (const float*)d_in[6];
  const float* enc_bhh  = (const float*)d_in[7];
  const float* encp_wih = (const float*)d_in[8];
  const float* encp_whh = (const float*)d_in[9];
  const float* encp_bih = (const float*)d_in[10];
  const float* encp_bhh = (const float*)d_in[11];
  const float* dec_wih  = (const float*)d_in[12];
  const float* dec_whh  = (const float*)d_in[13];
  const float* dec_bih  = (const float*)d_in[14];
  const float* dec_bhh  = (const float*)d_in[15];
  const float* lin_w    = (const float*)d_in[16];
  const float* lin_b    = (const float*)d_in[17];
  const float* pose_w   = (const float*)d_in[18];
  const float* pose_b   = (const float*)d_in[19];
  float* out = (float*)d_out;

  // ---- workspace bump allocator (256B aligned) ----
  char* base = (char*)d_ws;
  size_t off = 0;
  auto alloc = [&](size_t elems, size_t esz) -> void* {
    off = (off + 255) & ~(size_t)255;
    void* r = base + off;
    off += elems * esz;
    return r;
  };
  bf16_t* wf_pad      = (bf16_t*)alloc((size_t)D_DIM * KP_DIM, 2);
  bf16_t* enc_wih_bf  = (bf16_t*)alloc((size_t)FOURH * D_DIM, 2);
  bf16_t* enc_whh_bf  = (bf16_t*)alloc((size_t)FOURH * H_DIM, 2);
  bf16_t* encp_wih_bf = (bf16_t*)alloc((size_t)FOURH * D_DIM, 2);
  bf16_t* encp_whh_bf = (bf16_t*)alloc((size_t)FOURH * H_DIM, 2);
  bf16_t* dec_wih_bf  = (bf16_t*)alloc((size_t)FOURH * D_DIM, 2);
  bf16_t* dec_whh_bf  = (bf16_t*)alloc((size_t)FOURH * H_DIM, 2);
  bf16_t* lin_w_bf    = (bf16_t*)alloc((size_t)D_DIM * H_DIM, 2);
  bf16_t* pose_w_pad  = (bf16_t*)alloc((size_t)POSE_NPAD * H_DIM, 2);
  bf16_t* x_pad       = (bf16_t*)alloc((size_t)B_DIM * T_IN * KP_DIM, 2);
  bf16_t* z_pad       = (bf16_t*)alloc((size_t)B_DIM * T_IN * KP_DIM, 2);
  bf16_t* xf_bf       = (bf16_t*)alloc((size_t)T_IN * B_DIM * D_DIM, 2);
  bf16_t* zf_bf       = (bf16_t*)alloc((size_t)T_IN * B_DIM * D_DIM, 2);
  bf16_t* all_h_bf    = (bf16_t*)alloc((size_t)T_IN * B_DIM * H_DIM, 2);
  bf16_t* all_hz_bf   = (bf16_t*)alloc((size_t)T_IN * B_DIM * H_DIM, 2);
  bf16_t* atth_bf     = (bf16_t*)alloc((size_t)B_DIM * H_DIM, 2);
  bf16_t* zmid_bf     = (bf16_t*)alloc((size_t)B_DIM * D_DIM, 2);
  bf16_t* hs_bf       = (bf16_t*)alloc((size_t)T_OUT * B_DIM * H_DIM, 2);
  float*  G           = (float*)alloc((size_t)B_DIM * FOURH, 4);
  float*  c_enc       = (float*)alloc((size_t)B_DIM * H_DIM, 4);   // becomes decoder c
  float*  c_encp      = (float*)alloc((size_t)B_DIM * H_DIM, 4);
  float*  bc_enc      = (float*)alloc(FOURH, 4);
  float*  bc_encp     = (float*)alloc(FOURH, 4);
  float*  bc_dec      = (float*)alloc(FOURH, 4);

  auto nb = [](long n) { return (unsigned)((n + 255) / 256); };
  auto gemm = [&](const bf16_t* A, const bf16_t* W, const float* bias, const float* add,
                  float* oF, bf16_t* oB, int M, int N, int K, int Nstore, int ldC,
                  int permI, int permO) {
    dim3 g((unsigned)(M / 128), (unsigned)(N / 128));
    gemm_bf16_kernel<<<g, 256, 0, stream>>>(A, W, bias, add, oF, oB,
                                            M, N, K, Nstore, ldC, permI, permO);
  };

  // ---- setup: bf16 conversion, padding, combined biases, zero states ----
  k_pad_cols<<<nb((long)B_DIM * T_IN * KP_DIM), 256, 0, stream>>>(x, x_pad, B_DIM * T_IN, P_DIM, KP_DIM);
  k_pad_cols<<<nb((long)B_DIM * T_IN * KP_DIM), 256, 0, stream>>>(z, z_pad, B_DIM * T_IN, P_DIM, KP_DIM);
  k_pad_cols<<<nb((long)D_DIM * KP_DIM), 256, 0, stream>>>(wf, wf_pad, D_DIM, P_DIM, KP_DIM);
  k_f2bf<<<nb((long)FOURH * D_DIM), 256, 0, stream>>>(enc_wih,  enc_wih_bf,  (long)FOURH * D_DIM);
  k_f2bf<<<nb((long)FOURH * H_DIM), 256, 0, stream>>>(enc_whh,  enc_whh_bf,  (long)FOURH * H_DIM);
  k_f2bf<<<nb((long)FOURH * D_DIM), 256, 0, stream>>>(encp_wih, encp_wih_bf, (long)FOURH * D_DIM);
  k_f2bf<<<nb((long)FOURH * H_DIM), 256, 0, stream>>>(encp_whh, encp_whh_bf, (long)FOURH * H_DIM);
  k_f2bf<<<nb((long)FOURH * D_DIM), 256, 0, stream>>>(dec_wih,  dec_wih_bf,  (long)FOURH * D_DIM);
  k_f2bf<<<nb((long)FOURH * H_DIM), 256, 0, stream>>>(dec_whh,  dec_whh_bf,  (long)FOURH * H_DIM);
  k_f2bf<<<nb((long)D_DIM * H_DIM), 256, 0, stream>>>(lin_w,    lin_w_bf,    (long)D_DIM * H_DIM);
  k_pad_rows<<<nb((long)POSE_NPAD * H_DIM), 256, 0, stream>>>(pose_w, pose_w_pad, P_DIM, POSE_NPAD, H_DIM);
  k_add<<<nb(FOURH), 256, 0, stream>>>(enc_bih,  enc_bhh,  bc_enc,  FOURH);
  k_add<<<nb(FOURH), 256, 0, stream>>>(encp_bih, encp_bhh, bc_encp, FOURH);
  k_add<<<nb(FOURH), 256, 0, stream>>>(dec_bih,  dec_bhh,  bc_dec,  FOURH);
  k_zero<<<nb((long)B_DIM * H_DIM), 256, 0, stream>>>(c_enc,  (long)B_DIM * H_DIM);
  k_zero<<<nb((long)B_DIM * H_DIM), 256, 0, stream>>>(c_encp, (long)B_DIM * H_DIM);

  // ---- ToFeature: [B*T,96] x [512,96]^T -> bf16 [T,B,512] (row permute b*T+t -> t*B+b) ----
  gemm(x_pad, wf_pad, bf_, nullptr, nullptr, xf_bf,
       B_DIM * T_IN, D_DIM, KP_DIM, D_DIM, D_DIM, T_IN, B_DIM);
  gemm(z_pad, wf_pad, bf_, nullptr, nullptr, zf_bf,
       B_DIM * T_IN, D_DIM, KP_DIM, D_DIM, D_DIM, T_IN, B_DIM);

  // ---- encoder LSTMs (10 sequential steps each) ----
  const int BH = B_DIM * H_DIM;
  for (int t = 0; t < T_IN; ++t) {
    gemm(xf_bf + (size_t)t * B_DIM * D_DIM, enc_wih_bf, bc_enc, nullptr, G, nullptr,
         B_DIM, FOURH, D_DIM, FOURH, FOURH, 0, 0);
    if (t > 0)
      gemm(all_h_bf + (size_t)(t - 1) * BH, enc_whh_bf, nullptr, G, G, nullptr,
           B_DIM, FOURH, H_DIM, FOURH, FOURH, 0, 0);
    lstm_cell_kernel<<<nb(BH), 256, 0, stream>>>(G, c_enc, all_h_bf + (size_t)t * BH);
  }
  for (int t = 0; t < T_IN; ++t) {
    gemm(zf_bf + (size_t)t * B_DIM * D_DIM, encp_wih_bf, bc_encp, nullptr, G, nullptr,
         B_DIM, FOURH, D_DIM, FOURH, FOURH, 0, 0);
    if (t > 0)
      gemm(all_hz_bf + (size_t)(t - 1) * BH, encp_whh_bf, nullptr, G, G, nullptr,
           B_DIM, FOURH, H_DIM, FOURH, FOURH, 0, 0);
    lstm_cell_kernel<<<nb(BH), 256, 0, stream>>>(G, c_encp, all_hz_bf + (size_t)t * BH);
  }

  // ---- decoder: carry (h,c) = (enc last h, enc last c) ----
  const bf16_t* h_dec = all_h_bf + (size_t)(T_IN - 1) * BH;
  float* c_dec = c_enc;
  for (int t = 0; t < T_OUT; ++t) {
    // zero_mid = h @ lin_w^T + lin_b
    gemm(h_dec, lin_w_bf, lin_b, nullptr, nullptr, zmid_bf,
         B_DIM, D_DIM, H_DIM, D_DIM, D_DIM, 0, 0);
    // attention over cat([all_h, h, all_hz]) with query c
    attention_kernel<<<B_DIM, 256, 0, stream>>>(all_h_bf, h_dec, all_hz_bf, c_dec, atth_bf);
    // gates = zero_mid @ dec_wih^T + bias + atth @ dec_whh^T
    gemm(zmid_bf, dec_wih_bf, bc_dec, nullptr, G, nullptr,
         B_DIM, FOURH, D_DIM, FOURH, FOURH, 0, 0);
    gemm(atth_bf, dec_whh_bf, nullptr, G, G, nullptr,
         B_DIM, FOURH, H_DIM, FOURH, FOURH, 0, 0);
    lstm_cell_kernel<<<nb(BH), 256, 0, stream>>>(G, c_dec, hs_bf + (size_t)t * BH);
    h_dec = hs_bf + (size_t)t * BH;
  }

  // ---- ToPose: [25*B,1024] x [128(pad 66),1024]^T -> fp32 d_out [B,25,66]
  //      row permute t*B+b -> b*25+t, store only n < 66 ----
  gemm(hs_bf, pose_w_pad, pose_b, nullptr, out, nullptr,
       T_OUT * B_DIM, POSE_NPAD, H_DIM, P_DIM, P_DIM, B_DIM, T_OUT);
}
